// Model_3092376453869
// MI455X (gfx1250) — matrix operand order repaired and checked by reading, not compiled, not run on hardware
//
#include <hip/hip_runtime.h>

// ---------------------------------------------------------------------------
// Problem constants (match the reference)
// ---------------------------------------------------------------------------
#define T_STEPS 4096
#define IN_SZ   1024
#define HID     2048
#define HID4    8192   // 4*HID
#define MIDW    1024   // HID/2
#define OUT_SZ  256

typedef unsigned short ushort_t;
typedef __attribute__((ext_vector_type(16))) __bf16    v16bf;
typedef __attribute__((ext_vector_type(8)))  float     v8f;
typedef __attribute__((ext_vector_type(8)))  unsigned short u16x8;

union BFrag { u16x8 u[2]; v16bf v; };

// float -> bf16 (round to nearest even)
static __device__ __forceinline__ ushort_t f2bf(float f) {
    unsigned u = __float_as_uint(f);
    unsigned r = 0x7FFFu + ((u >> 16) & 1u);
    return (ushort_t)((u + r) >> 16);
}
static __device__ __forceinline__ float bf2f(ushort_t x) {
    return __uint_as_float(((unsigned)x) << 16);
}
static __device__ __forceinline__ float sigmoidf_(float x) {
    return 1.0f / (1.0f + expf(-x));
}

// ---------------------------------------------------------------------------
// Elementwise fp32 -> bf16 conversion
// ---------------------------------------------------------------------------
__global__ void cvt_bf16_kernel(const float* __restrict__ src,
                                ushort_t* __restrict__ dst, size_t n) {
    size_t i = (size_t)blockIdx.x * blockDim.x + threadIdx.x;
    size_t stride = (size_t)gridDim.x * blockDim.x;
    for (; i < n; i += stride) dst[i] = f2bf(src[i]);
}

// ---------------------------------------------------------------------------
// Pack W_hh [8192, 2048] fp32 -> bf16, interleaved so that the persistent
// mat-vec loop does fully coalesced 16B-per-lane loads.
// Layout: Wp[(k8*8192 + p)*8 + kk] = bf16(W_hh[r(p)][k8*8+kk])
//   p = blk*256 + tid ; tid: gi = tid/64 (gate), ul = tid%64 (unit-in-block)
//   r = gi*2048 + blk*64 + ul
// ---------------------------------------------------------------------------
__global__ void pack_whh_kernel(const float* __restrict__ Whh,
                                ushort_t* __restrict__ Wp) {
    unsigned idx = blockIdx.x * blockDim.x + threadIdx.x;   // one 8-group each
    if (idx >= 256u * 8192u) return;
    unsigned k8 = idx >> 13;          // 0..255
    unsigned p  = idx & 8191u;        // 0..8191
    unsigned b  = p >> 8;             // block 0..31
    unsigned t  = p & 255u;
    unsigned gi = t >> 6;             // gate 0..3
    unsigned ul = t & 63u;            // unit in block
    unsigned r  = gi * HID + b * 64u + ul;
    const float* src = Whh + (size_t)r * HID + (size_t)k8 * 8;
    ushort_t* d = Wp + (size_t)idx * 8;
#pragma unroll
    for (int kk = 0; kk < 8; ++kk) d[kk] = f2bf(src[kk]);
}

// ---------------------------------------------------------------------------
// Init: zero h_0 row of Hbuf, zero grid-barrier counter (re-done every call)
// ---------------------------------------------------------------------------
__global__ void init_state_kernel(float* __restrict__ Hbuf0,
                                  unsigned* __restrict__ cnt) {
    int i = blockIdx.x * blockDim.x + threadIdx.x;
    if (i < HID) Hbuf0[i] = 0.0f;
    if (i == 0)  *cnt = 0u;
}

// ---------------------------------------------------------------------------
// WMMA GEMM:  Cout[M,N] = act( A[M,K](bf16) @ W[N,K]^T(bf16) + bias )
// One wave -> 32x64 tile (2x4 frags of 16x16): 8 f32 accum frags (64 VGPRs)
// + 2 A frags + 4 B frags (48 VGPRs) => no scratch spills.
// Block = 256 threads (8 waves, 2 rows x 4 cols) -> block tile 64 x 256.
// grid = (N/256, M/64).
// EPI: 0 = +bias0+bias1 -> f32     (gate pre-activations GX)
//      1 = +bias0, relu -> bf16    (MID)
//      2 = +bias0       -> f32     (final output)
// ---------------------------------------------------------------------------
template <int EPI>
__global__ __launch_bounds__(256, 1)
void wmma_gemm_kernel(const ushort_t* __restrict__ A,
                      const ushort_t* __restrict__ W,
                      const float* __restrict__ bias0,
                      const float* __restrict__ bias1,
                      float* __restrict__ outF,
                      ushort_t* __restrict__ outBF,
                      int M, int N, int K) {
    const int lane = threadIdx.x & 31;
    const int wave = threadIdx.x >> 5;
    const int wm   = wave & 1;      // wave row (0..1)
    const int wn   = wave >> 1;     // wave col (0..3)
    const int half = lane >> 4;     // 0: lanes 0-15, 1: lanes 16-31
    const int l16  = lane & 15;

    const int m0 = blockIdx.y * 64 + wm * 32;
    const int n0 = blockIdx.x * 256 + wn * 64;

    v8f acc[2][4];
    const v8f vzero = {0.f, 0.f, 0.f, 0.f, 0.f, 0.f, 0.f, 0.f};
#pragma unroll
    for (int i = 0; i < 2; ++i)
#pragma unroll
        for (int j = 0; j < 4; ++j) acc[i][j] = vzero;

    // Hoisted per-lane row pointers; advanced by += 32 per k-step so only a
    // handful of 64-bit addresses stay live (no per-iter 64-bit recompute).
    const ushort_t* ap0 = A + (size_t)(m0 + l16) * K + half * 16;
    const ushort_t* ap1 = A + (size_t)(m0 + 16 + l16) * K + half * 16;
    const ushort_t* bp0 = W + (size_t)(n0 + l16) * K + half * 16;
    const ushort_t* bp1 = W + (size_t)(n0 + 16 + l16) * K + half * 16;
    const ushort_t* bp2 = W + (size_t)(n0 + 32 + l16) * K + half * 16;
    const ushort_t* bp3 = W + (size_t)(n0 + 48 + l16) * K + half * 16;

#pragma unroll 1
    for (int k0 = 0; k0 < K; k0 += 32) {
        BFrag a[2], b[4];
        // A fragment (16x32 bf16): lane<16 elems 0..15 = K[+0..15],
        // lane>=16 elems 0..15 = K[+16..31] (same k order as B).
        a[0].u[0] = *(const u16x8*)(ap0);
        a[0].u[1] = *(const u16x8*)(ap0 + 8);
        a[1].u[0] = *(const u16x8*)(ap1);
        a[1].u[1] = *(const u16x8*)(ap1 + 8);
        // B fragment (32x16 bf16): B[k][n] = W[n][k] -> lane<16 reads
        // W[n0+..][+0..15]; lane>=16 reads [+16..31] (contiguous in W rows).
        b[0].u[0] = *(const u16x8*)(bp0);
        b[0].u[1] = *(const u16x8*)(bp0 + 8);
        b[1].u[0] = *(const u16x8*)(bp1);
        b[1].u[1] = *(const u16x8*)(bp1 + 8);
        b[2].u[0] = *(const u16x8*)(bp2);
        b[2].u[1] = *(const u16x8*)(bp2 + 8);
        b[3].u[0] = *(const u16x8*)(bp3);
        b[3].u[1] = *(const u16x8*)(bp3 + 8);

#pragma unroll
        for (int i = 0; i < 2; ++i)
#pragma unroll
            for (int j = 0; j < 4; ++j)
                acc[i][j] = __builtin_amdgcn_wmma_f32_16x16x32_bf16(
                    false, a[i].v, false, b[j].v, (short)0, acc[i][j],
                    false, false);

        ap0 += 32; ap1 += 32;
        bp0 += 32; bp1 += 32; bp2 += 32; bp3 += 32;
    }

    // Epilogue. C/D layout: VGPR r -> row m0+16i + half*8 + r, col n0+16j+l16.
#pragma unroll
    for (int j = 0; j < 4; ++j) {
        const int col = n0 + j * 16 + l16;
        float bv = bias0[col];
        if (EPI == 0) bv += bias1[col];
#pragma unroll
        for (int i = 0; i < 2; ++i) {
            const int rowbase = m0 + i * 16 + half * 8;
#pragma unroll
            for (int r = 0; r < 8; ++r) {
                float v = acc[i][j][r] + bv;
                const size_t off = (size_t)(rowbase + r) * N + col;
                if (EPI == 1) {
                    v = fmaxf(v, 0.0f);
                    outBF[off] = f2bf(v);
                } else {
                    outF[off] = v;
                }
            }
        }
    }
}

// ---------------------------------------------------------------------------
// Persistent LSTM recurrence. 32 blocks x 256 threads = 8192 threads;
// thread p computes one gate row dot(Whh[r], h) per step from the packed
// bf16 weights (L2-resident, coalesced), gates exchanged via LDS, c-state in
// registers, one monotonic-counter grid barrier per step.
// ---------------------------------------------------------------------------
__global__ __launch_bounds__(256, 1)
void lstm_persistent_kernel(const ushort_t* __restrict__ Wp,
                            const float* __restrict__ GX,
                            float* __restrict__ Hbuf,   // (T+1) x HID
                            ushort_t* __restrict__ Hbf, // T x HID
                            unsigned* __restrict__ cnt,
                            int nsteps) {
    __shared__ float h_lds[HID];
    __shared__ float g_lds[256];

    const int tid = threadIdx.x;
    const int bid = blockIdx.x;
    const int G   = gridDim.x;
    const int p   = bid * 256 + tid;        // 0..8191
    const int gi  = tid >> 6;               // gate index 0..3
    const int ul  = tid & 63;               // unit within block
    const int u   = bid * 64 + ul;          // hidden unit (for gi==0 threads)
    const int r   = gi * HID + u;           // gate-matrix row owned by thread

    const ushort_t* wbase = Wp + (size_t)p * 8;

    float c = 0.0f;

    for (int t = 0; t < nsteps; ++t) {
        // stage h_t into LDS
        for (int i = tid; i < HID; i += 256)
            h_lds[i] = Hbuf[(size_t)t * HID + i];
        __syncthreads();

        // dot(Whh_bf16[r], h)  -- 256 x 16B coalesced loads per thread
        float acc = 0.0f;
#pragma unroll 2
        for (int k8 = 0; k8 < 256; ++k8) {
            u16x8 w = *(const u16x8*)(wbase + (size_t)k8 * (8192 * 8));
            if (k8 + 4 < 256)
                __builtin_prefetch(
                    (const void*)(wbase + (size_t)(k8 + 4) * (8192 * 8)), 0, 0);
            const float* hp = &h_lds[k8 * 8];
            acc += bf2f(w[0]) * hp[0] + bf2f(w[1]) * hp[1] +
                   bf2f(w[2]) * hp[2] + bf2f(w[3]) * hp[3] +
                   bf2f(w[4]) * hp[4] + bf2f(w[5]) * hp[5] +
                   bf2f(w[6]) * hp[6] + bf2f(w[7]) * hp[7];
        }
        float gate = acc + GX[(size_t)t * HID4 + r];

        g_lds[tid] = gate;      // index == gi*64 + ul
        __syncthreads();

        if (tid < 64) {         // gi==0 threads own units u = bid*64 + tid
            float iv = sigmoidf_(g_lds[tid]);
            float fv = sigmoidf_(g_lds[64 + tid]);
            float gv = tanhf(g_lds[128 + tid]);
            float ov = sigmoidf_(g_lds[192 + tid]);
            c = fv * c + iv * gv;
            float h = ov * tanhf(c);
            h = fmaxf(h, 0.0f);                       // reference ReLU on h
            Hbuf[(size_t)(t + 1) * HID + u] = h;      // next step's input
            Hbf[(size_t)t * HID + u] = f2bf(h);       // for GEMM2
        }

        // grid-wide barrier (monotonic counter; 32 blocks co-resident)
        __syncthreads();
        if (tid == 0) {
            __threadfence();
            atomicAdd(cnt, 1u);
            const unsigned goal = (unsigned)(t + 1) * (unsigned)G;
            while (__hip_atomic_load(cnt, __ATOMIC_ACQUIRE,
                                     __HIP_MEMORY_SCOPE_AGENT) < goal)
                __builtin_amdgcn_s_sleep(2);
        }
        __syncthreads();
    }
}

// ---------------------------------------------------------------------------
// Host-side orchestration
// ---------------------------------------------------------------------------
static inline size_t align256(size_t x) { return (x + 255) & ~(size_t)255; }

extern "C" void kernel_launch(void* const* d_in, const int* in_sizes, int n_in,
                              void* d_out, int out_size, void* d_ws,
                              size_t ws_size, hipStream_t stream) {
    const float* x    = (const float*)d_in[0];   // [T, 1024]
    const float* W_ih = (const float*)d_in[1];   // [8192, 1024]
    const float* W_hh = (const float*)d_in[2];   // [8192, 2048]
    const float* b_ih = (const float*)d_in[3];   // [8192]
    const float* b_hh = (const float*)d_in[4];   // [8192]
    const float* W2   = (const float*)d_in[5];   // [1024, 2048]
    const float* b2   = (const float*)d_in[6];   // [1024]
    const float* W3   = (const float*)d_in[7];   // [256, 1024]
    const float* b3   = (const float*)d_in[8];   // [256]
    float* out = (float*)d_out;                  // [T, 256]

    // ---- workspace carve-up ----
    char* ws = (char*)d_ws;
    size_t off = 0;
    float* GX = (float*)(ws + off);        off += align256((size_t)T_STEPS * HID4 * 4);
    float* Hbuf = (float*)(ws + off);      off += align256((size_t)(T_STEPS + 1) * HID * 4);
    ushort_t* Hbf = (ushort_t*)(ws + off); off += align256((size_t)T_STEPS * HID * 2);
    ushort_t* MIDbf = (ushort_t*)(ws + off); off += align256((size_t)T_STEPS * MIDW * 2);
    ushort_t* Xbf = (ushort_t*)(ws + off);   off += align256((size_t)T_STEPS * IN_SZ * 2);
    ushort_t* Wihbf = (ushort_t*)(ws + off); off += align256((size_t)HID4 * IN_SZ * 2);
    ushort_t* W2bf = (ushort_t*)(ws + off);  off += align256((size_t)MIDW * HID * 2);
    ushort_t* W3bf = (ushort_t*)(ws + off);  off += align256((size_t)OUT_SZ * MIDW * 2);
    ushort_t* Wp = (ushort_t*)(ws + off);    off += align256((size_t)HID4 * HID * 2);
    unsigned* cnt = (unsigned*)(ws + off);   off += 256;
    (void)ws_size; (void)n_in; (void)in_sizes; (void)out_size;

    // ---- 0. init h0 + barrier counter ----
    init_state_kernel<<<8, 256, 0, stream>>>(Hbuf, cnt);

    // ---- 1. fp32 -> bf16 conversions ----
    cvt_bf16_kernel<<<4096, 256, 0, stream>>>(x, Xbf, (size_t)T_STEPS * IN_SZ);
    cvt_bf16_kernel<<<4096, 256, 0, stream>>>(W_ih, Wihbf, (size_t)HID4 * IN_SZ);
    cvt_bf16_kernel<<<4096, 256, 0, stream>>>(W2, W2bf, (size_t)MIDW * HID);
    cvt_bf16_kernel<<<1024, 256, 0, stream>>>(W3, W3bf, (size_t)OUT_SZ * MIDW);
    pack_whh_kernel<<<8192, 256, 0, stream>>>(W_hh, Wp);

    // ---- 2. GX = X @ W_ih^T + (b_ih + b_hh)   [4096 x 8192] ----
    wmma_gemm_kernel<0><<<dim3(HID4 / 256, T_STEPS / 64), 256, 0, stream>>>(
        Xbf, Wihbf, b_ih, b_hh, GX, nullptr, T_STEPS, HID4, IN_SZ);

    // ---- 3. persistent LSTM recurrence ----
    lstm_persistent_kernel<<<32, 256, 0, stream>>>(Wp, GX, Hbuf, Hbf, cnt,
                                                   T_STEPS);

    // ---- 4. MID = relu(H @ W2^T + b2)  -> bf16  [4096 x 1024] ----
    wmma_gemm_kernel<1><<<dim3(MIDW / 256, T_STEPS / 64), 256, 0, stream>>>(
        Hbf, W2bf, b2, nullptr, nullptr, MIDbf, T_STEPS, MIDW, HID);

    // ---- 5. OUT = MID @ W3^T + b3  -> fp32  [4096 x 256] ----
    wmma_gemm_kernel<2><<<dim3(OUT_SZ / 256, T_STEPS / 64), 256, 0, stream>>>(
        MIDbf, W3bf, b3, nullptr, out, nullptr, T_STEPS, OUT_SZ, MIDW);
}
